// KANLayer_70385924047526
// MI455X (gfx1250) — matrix operand order repaired and checked by reading, not compiled, not run on hardware
//
#include <hip/hip_runtime.h>
#include <hip/hip_bf16.h>

// Problem constants (match reference)
#define B_TOTAL 65536
#define IN_DIM  256
#define G_DIM   512
#define WIDTH   512
#define LN_EPS  1e-5f

#define M_TILE   64          // batch rows per block
#define THREADS  256         // 8 waves of 32
#define XH_STRIDE 264        // halves per row (256 + 8 pad), 16B-aligned rows
#define SH_STRIDE 520        // halves per row (512 + 8 pad)
#define EH_STRIDE 520        // floats per row (512 + 8 pad)

typedef __attribute__((ext_vector_type(16))) _Float16 v16h;
typedef __attribute__((ext_vector_type(8)))  _Float16 v8h;
typedef __attribute__((ext_vector_type(4)))  _Float16 v4h;
typedef __attribute__((ext_vector_type(8)))  float    v8f;

__device__ __forceinline__ v8h cvt8(float4 a, float4 b) {
    v8h r;
    r[0] = (_Float16)a.x; r[1] = (_Float16)a.y;
    r[2] = (_Float16)a.z; r[3] = (_Float16)a.w;
    r[4] = (_Float16)b.x; r[5] = (_Float16)b.y;
    r[6] = (_Float16)b.z; r[7] = (_Float16)b.w;
    return r;
}

__device__ __forceinline__ v16h combine16(v8h lo, v8h hi) {
    v16h r;
#pragma unroll
    for (int i = 0; i < 8; ++i) { r[i] = lo[i]; r[i + 8] = hi[i]; }
    return r;
}

// B-fragment load: 16 contiguous K values for this lane's (N=nlane, K-half).
// F16B path: single 32-byte f16 load.  f32 path: 4x16B loads + pack-convert.
template <bool F16B>
__device__ __forceinline__ v16h load_bfrag(const float* __restrict__ rowf,
                                           const _Float16* __restrict__ rowh,
                                           int koff) {
    if constexpr (F16B) {
        return *(const v16h*)(rowh + koff);
    } else {
        const float4* cp = (const float4*)(rowf + koff);
        return combine16(cvt8(cp[0], cp[1]), cvt8(cp[2], cp[3]));
    }
}

// Pre-kernel: csq[g] = sum_k centers[g][k]^2
extern "C" __global__ void kan_pre(const float* __restrict__ C, float* __restrict__ csq) {
    int g = blockIdx.x * blockDim.x + threadIdx.x;
    if (g < G_DIM) {
        const float4* cp = (const float4*)(C + (size_t)g * IN_DIM);
        float s = 0.f;
#pragma unroll 8
        for (int i = 0; i < IN_DIM / 4; ++i) {
            float4 v = cp[i];
            s += v.x * v.x + v.y * v.y + v.z * v.z + v.w * v.w;
        }
        csq[g] = s;
    }
}

// Pre-kernel: f32 -> f16 conversion of a weight matrix (one float4 per thread)
extern "C" __global__ void kan_cvt_f16(const float* __restrict__ src,
                                       _Float16* __restrict__ dst, int n4) {
    int i = blockIdx.x * blockDim.x + threadIdx.x;
    if (i < n4) {
        float4 v = ((const float4*)src)[i];
        v4h h;
        h[0] = (_Float16)v.x; h[1] = (_Float16)v.y;
        h[2] = (_Float16)v.z; h[3] = (_Float16)v.w;
        *(v4h*)(dst + (size_t)i * 4) = h;
    }
}

template <bool F16B>
__global__ __launch_bounds__(THREADS, 1)
void kan_fused(const float* __restrict__ X,
               const float* __restrict__ Cg,
               const float* __restrict__ logs,
               const float* __restrict__ Wm,
               const float* __restrict__ bm,
               const float* __restrict__ gam,
               const float* __restrict__ bet,
               const float* __restrict__ csq,
               const _Float16* __restrict__ Ch,   // f16 centers (G x IN), may be null if !F16B
               const _Float16* __restrict__ Wh,   // f16 W_mix  (WIDTH x G), may be null if !F16B
               float* __restrict__ out)
{
    // Single LDS buffer, phases alias it:
    //   [phase0..1] Xh (f16, 64x264) at byte 66560;  Sh (f16, 64x520) at byte 0
    //   [phase3]    Eh (f32, 64x520) spans the whole buffer (Xh/Sh dead by then)
    __shared__ __align__(16) unsigned char smem[M_TILE * EH_STRIDE * 4];
    __shared__ float xsq_s[M_TILE];

    _Float16* Sh = (_Float16*)smem;
    _Float16* Xh = (_Float16*)(smem + M_TILE * SH_STRIDE * 2);
    float*    Eh = (float*)smem;

    const int t    = threadIdx.x;
    const int lane = t & 31;
    const int wave = t >> 5;
    const int mg   = wave & 3;        // 16-row M group (0..3)
    const int nh   = wave >> 2;       // which 256-wide N half (0..1)
    const int row0 = blockIdx.x * M_TILE;

    const int nlane = lane & 15;              // N within a 16-col tile
    const int khalf = (lane >> 4) ? 16 : 0;   // B-operand K half per lane-half
    const int asel  = (lane >> 4) ? 16 : 0;   // A-operand K sub-offset per lane-half
    const int mbase = mg * 16 + ((lane >> 4) << 3);  // D-operand M base per lane-half

    // ---------------- Phase 0: X tile -> f16 LDS + ||x||^2 ----------------
    {
        const int r = t >> 2;     // row 0..63
        const int q = t & 3;      // quarter of the 256 input dims
        const float4* xp = (const float4*)(X + (size_t)(row0 + r) * IN_DIM + q * 64);
        _Float16* xh = Xh + r * XH_STRIDE + q * 64;
        float s = 0.f;
#pragma unroll
        for (int i = 0; i < 16; ++i) {
            float4 v = xp[i];
            s += v.x * v.x + v.y * v.y + v.z * v.z + v.w * v.w;
            xh[i * 4 + 0] = (_Float16)v.x;
            xh[i * 4 + 1] = (_Float16)v.y;
            xh[i * 4 + 2] = (_Float16)v.z;
            xh[i * 4 + 3] = (_Float16)v.w;
        }
        // butterfly over the 4 consecutive lanes sharing a row
        s += __shfl_xor(s, 1);
        s += __shfl_xor(s, 2);
        if (q == 0) xsq_s[r] = s;
    }
    __syncthreads();

    // ---------------- Phase 1: GEMM1 (X . C^T) + RBF -> Sh ----------------
    // A-fragments for this wave's 16 rows, all K=256, held in registers.
    v16h afrag[8];
    {
        const _Float16* xr = Xh + (mg * 16 + nlane) * XH_STRIDE;
#pragma unroll
        for (int kc = 0; kc < 8; ++kc) {
            v8h lo = *(const v8h*)(xr + kc * 32 + asel);
            v8h hi = *(const v8h*)(xr + kc * 32 + 8 + asel);
            afrag[kc] = combine16(lo, hi);
        }
    }

#pragma unroll 1
    for (int nt = 0; nt < 16; ++nt) {
        const int n = nh * 256 + nt * 16 + nlane;
        const float*    crowf = Cg + (size_t)n * IN_DIM + khalf;
        const _Float16* crowh = F16B ? (Ch + (size_t)n * IN_DIM + khalf) : nullptr;
        if (nt < 15) {
            // warm next tile's center row toward the WGP (global_prefetch_b8)
            if constexpr (F16B) __builtin_prefetch(crowh + 16 * IN_DIM, 0, 0);
            else                __builtin_prefetch(crowf + 16 * IN_DIM, 0, 0);
        }
        v8f acc = {};
#pragma unroll
        for (int kc = 0; kc < 8; ++kc) {
            v16h bfrag = load_bfrag<F16B>(crowf, crowh, kc * 32);
            acc = __builtin_amdgcn_wmma_f32_16x16x32_f16(
                false, afrag[kc], false, bfrag, (short)0, acc, false, false);
        }
        const float cs = csq[n];
        const float sc = __expf(logs[n]);
#pragma unroll
        for (int r = 0; r < 8; ++r) {
            const int m = mbase + r;
            float d2 = xsq_s[m] + cs - 2.0f * acc[r];
            float sv = __expf(-d2 * sc);
            Sh[m * SH_STRIDE + nh * 256 + nt * 16 + nlane] = (_Float16)sv;
        }
    }
    __syncthreads();

    // ---------------- Phase 2: GEMM2 A-fragments from Sh -> registers ------
    v16h a2[16];
    {
        const _Float16* sr = Sh + (mg * 16 + nlane) * SH_STRIDE;
#pragma unroll
        for (int kc = 0; kc < 16; ++kc) {
            v8h lo = *(const v8h*)(sr + kc * 32 + asel);
            v8h hi = *(const v8h*)(sr + kc * 32 + 8 + asel);
            a2[kc] = combine16(lo, hi);
        }
    }
    __syncthreads();   // Sh fully consumed; Eh may now overwrite the buffer

    // ---------------- Phase 3: GEMM2 (scaled . W^T) + bias -> Eh -----------
#pragma unroll 1
    for (int nt = 0; nt < 16; ++nt) {
        const int w = nh * 256 + nt * 16 + nlane;
        const float*    wrowf = Wm + (size_t)w * G_DIM + khalf;
        const _Float16* wrowh = F16B ? (Wh + (size_t)w * G_DIM + khalf) : nullptr;
        if (nt < 15) {
            if constexpr (F16B) __builtin_prefetch(wrowh + 16 * G_DIM, 0, 0);
            else                __builtin_prefetch(wrowf + 16 * G_DIM, 0, 0);
        }
        v8f acc = {};
#pragma unroll
        for (int kc = 0; kc < 16; ++kc) {
            v16h bfrag = load_bfrag<F16B>(wrowf, wrowh, kc * 32);
            acc = __builtin_amdgcn_wmma_f32_16x16x32_f16(
                false, a2[kc], false, bfrag, (short)0, acc, false, false);
        }
        const float bias = bm[w];
#pragma unroll
        for (int r = 0; r < 8; ++r)
            Eh[(mbase + r) * EH_STRIDE + w] = acc[r] + bias;
    }
    __syncthreads();

    // ---------------- Phase 4: LayerNorm + tanh -> out ---------------------
    {
        const int r = t >> 2;
        const int q = t & 3;
        const float* er = Eh + r * EH_STRIDE + q * 128;
        float s = 0.f, s2 = 0.f;
#pragma unroll 8
        for (int i = 0; i < 128; ++i) {
            float v = er[i];
            s += v; s2 += v * v;
        }
        s  += __shfl_xor(s, 1);  s  += __shfl_xor(s, 2);
        s2 += __shfl_xor(s2, 1); s2 += __shfl_xor(s2, 2);
        const float mean = s * (1.0f / WIDTH);
        float var = s2 * (1.0f / WIDTH) - mean * mean;
        const float inv = rsqrtf(var + LN_EPS);

        float* orow = out + (size_t)(row0 + r) * WIDTH + q * 128;
        const float* gq = gam + q * 128;
        const float* bq = bet + q * 128;
#pragma unroll 4
        for (int i = 0; i < 128; ++i) {
            float nv = (er[i] - mean) * inv;
            orow[i] = tanhf(nv * gq[i] + bq[i]);
        }
    }
}

extern "C" void kernel_launch(void* const* d_in, const int* in_sizes, int n_in,
                              void* d_out, int out_size, void* d_ws, size_t ws_size,
                              hipStream_t stream) {
    const float* X    = (const float*)d_in[0];   // (B, IN)
    const float* Cg   = (const float*)d_in[1];   // (G, IN)
    const float* logs = (const float*)d_in[2];   // (G,)
    const float* Wm   = (const float*)d_in[3];   // (WIDTH, G)
    const float* bm   = (const float*)d_in[4];   // (WIDTH,)
    const float* gam  = (const float*)d_in[5];   // (WIDTH,)
    const float* bet  = (const float*)d_in[6];   // (WIDTH,)
    float* out = (float*)d_out;                  // (B, WIDTH)

    // Workspace layout: csq (2KB, padded) | Ch f16 (256KB) | Wh f16 (512KB)
    float* csq = (float*)d_ws;
    const size_t need = 2048 + (size_t)G_DIM * IN_DIM * 2 + (size_t)WIDTH * G_DIM * 2;

    kan_pre<<<(G_DIM + 255) / 256, 256, 0, stream>>>(Cg, csq);

    if (ws_size >= need) {
        _Float16* Ch = (_Float16*)((char*)d_ws + 2048);
        _Float16* Wh = Ch + (size_t)G_DIM * IN_DIM;
        const int nC4 = G_DIM * IN_DIM / 4;    // 32768
        const int nW4 = WIDTH * G_DIM / 4;     // 65536
        kan_cvt_f16<<<(nC4 + 255) / 256, 256, 0, stream>>>(Cg, Ch, nC4);
        kan_cvt_f16<<<(nW4 + 255) / 256, 256, 0, stream>>>(Wm, Wh, nW4);
        kan_fused<true><<<B_TOTAL / M_TILE, THREADS, 0, stream>>>(
            X, Cg, logs, Wm, bm, gam, bet, csq, Ch, Wh, out);
    } else {
        kan_fused<false><<<B_TOTAL / M_TILE, THREADS, 0, stream>>>(
            X, Cg, logs, Wm, bm, gam, bet, csq, nullptr, nullptr, out);
    }
}
